// BitLinearPacked_1958505087323
// MI455X (gfx1250) — matrix operand order repaired and checked by reading, not compiled, not run on hardware
//
#include <hip/hip_runtime.h>
#include <hip/hip_bf16.h>
#include <stdint.h>

// ---------------------------------------------------------------------------
// BitLinearPacked: out[b,s,o] = sum_i x[b,s,i] * w[o,i] * scale
//   x : [4,2048,4096] f32   -> M = 8192, K = 4096
//   bp: [4096*4096/8] int32 (one byte-value per element, MSB-first bits)
//   w[o,i] = bit ? +1 : -1   (exact in bf16)
// Strategy (MI455X / gfx1250, wave32):
//   pass1: x f32 -> bf16 (RNE) into d_ws (64 MiB)
//   pass2: WMMA bf16 GEMM, BM=128 BN=128 BK=64, 8 waves/block.
//          A tile: global->LDS via async-to-LDS (ASYNCcnt), double buffered.
//          B tile: packed bits loaded b128, unpacked to bf16 +-1 in VALU,
//                  stored to LDS (never materialized in HBM).
// ---------------------------------------------------------------------------

#define OUT_F 4096
#define IN_F  4096
#define BM 128
#define BN 128
#define BK 64
#define USE_ASYNC_COPY 1

typedef __attribute__((ext_vector_type(16))) __bf16 v16bf;
typedef __attribute__((ext_vector_type(8)))  float  v8f;

union FragU { uint4 u[2]; v16bf v; };

// ------------------------- pass 1: f32 -> bf16 (RNE) -----------------------
__device__ __forceinline__ uint32_t f2bf_rne(float f) {
  uint32_t u = __builtin_bit_cast(uint32_t, f);
  return (u + 0x7FFFu + ((u >> 16) & 1u)) >> 16;
}

__global__ __launch_bounds__(256) void cvt_x_bf16_kernel(
    const float* __restrict__ x, uint16_t* __restrict__ xb, int n) {
  int i = (blockIdx.x * 256 + threadIdx.x) * 8;
  if (i >= n) return;
  float4 a = *(const float4*)(x + i);
  float4 b = *(const float4*)(x + i + 4);
  uint4 o;
  o.x = f2bf_rne(a.x) | (f2bf_rne(a.y) << 16);
  o.y = f2bf_rne(a.z) | (f2bf_rne(a.w) << 16);
  o.z = f2bf_rne(b.x) | (f2bf_rne(b.y) << 16);
  o.w = f2bf_rne(b.z) | (f2bf_rne(b.w) << 16);
  *(uint4*)(xb + i) = o;
}

// ------------------------- pass 2: WMMA GEMM -------------------------------
// A tile loader: 128 rows x 64 bf16 = 16 KiB, 256 threads x 4 x b128.
__device__ __forceinline__ void load_A_tile(const uint16_t* __restrict__ g0,
                                            uint16_t* lds) {
  const int tid = threadIdx.x;
#pragma unroll
  for (int c = 0; c < 4; ++c) {
    int id  = c * 256 + tid;
    int row = id >> 3;        // 8 chunks of 16B per 128B row
    int cc  = id & 7;
    const uint16_t* g = g0 + row * IN_F + cc * 8;
#if USE_ASYNC_COPY
    uint32_t laddr = (uint32_t)(uintptr_t)(lds + row * BK + cc * 8);
    asm volatile("global_load_async_to_lds_b128 %0, %1, off"
                 :: "v"(laddr), "v"(g) : "memory");
#else
    *(uint4*)(lds + row * BK + cc * 8) = *(const uint4*)g;
#endif
  }
}

// B tile loader: 128 weight rows x 64 bits. Each thread: 4 packed ints
// (int4 load) -> 32 bf16 (+1/-1) -> 4 x ds_store_b128.
__device__ __forceinline__ void load_B_tile(const int* __restrict__ g0,
                                            uint16_t* lds) {
  const int tid  = threadIdx.x;
  const int n    = tid >> 1;        // 0..127  (weight row within tile)
  const int half = tid & 1;         // which 32 k-values of the 64
  int4 v = *(const int4*)(g0 + n * (IN_F / 8) + half * 4);
  int bytes[4] = {v.x, v.y, v.z, v.w};
  uint16_t* dst = lds + n * BK + half * 32;
#pragma unroll
  for (int q = 0; q < 4; ++q) {
    uint32_t by = (uint32_t)bytes[q];
    uint32_t w[4];
#pragma unroll
    for (int p = 0; p < 4; ++p) {
      // k = q*8 + 2p (lo half), q*8 + 2p+1 (hi half); MSB-first bit order.
      uint32_t b0 = (by >> (7 - 2 * p)) & 1u;
      uint32_t b1 = (by >> (6 - 2 * p)) & 1u;
      w[p] = (b0 ? 0x3F80u : 0xBF80u) | ((b1 ? 0x3F80u : 0xBF80u) << 16);
    }
    *(uint4*)(dst + q * 8) = make_uint4(w[0], w[1], w[2], w[3]);
  }
}

__global__ __launch_bounds__(256) void bitlinear_wmma_kernel(
    const uint16_t* __restrict__ xb,   // [M][K] bf16
    const int* __restrict__ bp,        // [N * K/8] one byte-value per int
    const float* __restrict__ scale,   // [1]
    float* __restrict__ out,           // [M][N]
    int M) {
  __shared__ __align__(16) uint16_t As[2][BM * BK];   // 2 x 16 KiB
  __shared__ __align__(16) uint16_t Bs[2][BN * BK];   // 2 x 16 KiB

  const int tid  = threadIdx.x;
  const int lane = tid & 31;
  const int wave = tid >> 5;
  const int wm   = wave & 3;              // 4 waves along M
  const int wn   = wave >> 2;             // 2 waves along N
  const int m0   = blockIdx.y * BM;
  const int n0   = blockIdx.x * BN;
  const int KT   = IN_F / BK;             // 64 K-tiles

  const uint16_t* gA = xb + (size_t)m0 * IN_F;
  const int*      gB = bp + (size_t)n0 * (IN_F / 8);

  v8f acc[2][4];
#pragma unroll
  for (int i = 0; i < 2; ++i)
#pragma unroll
    for (int j = 0; j < 4; ++j)
      acc[i][j] = (v8f){0.f, 0.f, 0.f, 0.f, 0.f, 0.f, 0.f, 0.f};

  // Preload tile 0 into buffer 0.
  load_A_tile(gA, As[0]);
  load_B_tile(gB, Bs[0]);

  const int lo16 = lane & 15;
  const int hi   = lane >> 4;            // 0 | 1

#pragma unroll 1
  for (int kt = 0; kt < KT; ++kt) {
    const int cur = kt & 1;
#if USE_ASYNC_COPY
    asm volatile("s_wait_asynccnt 0x0" ::: "memory");
#endif
    __syncthreads();                     // lds[cur] populated, lds[1-cur] free

    if (kt + 1 < KT) {                   // prefetch next tile into other buf
      load_A_tile(gA + (kt + 1) * BK, As[cur ^ 1]);
      load_B_tile(gB + (kt + 1) * (BK / 8), Bs[cur ^ 1]);
    }

#pragma unroll
    for (int ks = 0; ks < 2; ++ks) {     // two 16x16x32 steps per BK=64
      const int kk = ks * 32;
      // A fragments: ISA 16-bit A layout. lane<16 -> K {0..7,16..23},
      // lane>=16 -> K {8..15,24..31}, row = wm*32 + i*16 + (lane&15).
      v16bf afrag[2];
      const int aK = kk + (hi << 3);
#pragma unroll
      for (int i = 0; i < 2; ++i) {
        const uint16_t* p = &As[cur][(wm * 32 + i * 16 + lo16) * BK + aK];
        FragU f;
        f.u[0] = *(const uint4*)p;        // K .. K+7
        f.u[1] = *(const uint4*)(p + 16); // K+16 .. K+23
        afrag[i] = f.v;
      }
      // B fragments: col = wn*64 + j*16 + (lane&15);
      // lane<16 -> K {0..7,16..23}, lane>=16 -> K {8..15,24..31} (same as A).
      v16bf bfrag[4];
      const int bK = kk + (hi << 3);
#pragma unroll
      for (int j = 0; j < 4; ++j) {
        const uint16_t* p = &Bs[cur][(wn * 64 + j * 16 + lo16) * BK + bK];
        FragU f;
        f.u[0] = *(const uint4*)p;
        f.u[1] = *(const uint4*)(p + 16);
        bfrag[j] = f.v;
      }
#pragma unroll
      for (int i = 0; i < 2; ++i)
#pragma unroll
        for (int j = 0; j < 4; ++j)
          acc[i][j] = __builtin_amdgcn_wmma_f32_16x16x32_bf16(
              false, afrag[i], false, bfrag[j],
              (short)0, acc[i][j], false, false);
    }
  }

  // Epilogue: C/D layout -> row = r + (lane>=16 ? 8 : 0), col = lane&15.
  const float s = scale[0];
  const int rowBase = m0 + wm * 32 + hi * 8;
  const int colBase = n0 + wn * 64 + lo16;
#pragma unroll
  for (int i = 0; i < 2; ++i)
#pragma unroll
    for (int j = 0; j < 4; ++j)
#pragma unroll
      for (int r = 0; r < 8; ++r)
        out[(size_t)(rowBase + i * 16 + r) * OUT_F + (colBase + j * 16)] =
            acc[i][j][r] * s;
  (void)M;
}

// ---------------------------------------------------------------------------
extern "C" void kernel_launch(void* const* d_in, const int* in_sizes, int n_in,
                              void* d_out, int out_size, void* d_ws, size_t ws_size,
                              hipStream_t stream) {
  const float* x     = (const float*)d_in[0];
  const int*   bp    = (const int*)d_in[1];
  const float* scale = (const float*)d_in[2];
  float*       out   = (float*)d_out;

  const int nx = in_sizes[0];          // 8192*4096 = 33,554,432
  const int M  = nx / IN_F;            // 8192
  uint16_t* xb = (uint16_t*)d_ws;      // needs 64 MiB of workspace

  cvt_x_bf16_kernel<<<nx / (8 * 256), 256, 0, stream>>>(x, xb, nx);

  dim3 grid(OUT_F / BN, M / BM);       // 32 x 64 blocks
  bitlinear_wmma_kernel<<<grid, 256, 0, stream>>>(xb, bp, scale, out, M);
  (void)n_in; (void)out_size; (void)ws_size;
}
